// HGTPredictor_27685359190071
// MI455X (gfx1250) — matrix operand order repaired and checked by reading, not compiled, not run on hardware
//
#include <hip/hip_runtime.h>
#include <cstdint>
#include <cstddef>

#define NCn 50000
#define NDn 20000
#define NGn 30000
#define Dn  128
#define Hn  4
#define DHn 32
#define En  200000
#define Pn  200000
#define An  10
#define Sn  5
#define EADn 32
#define Ln  2

typedef __attribute__((ext_vector_type(16))) _Float16 v16h;
typedef __attribute__((ext_vector_type(8)))  _Float16 v8h;
typedef __attribute__((ext_vector_type(8)))  float    v8f;

// ---------- helpers ----------
__device__ __forceinline__ unsigned f2ord(float f) {
  unsigned b = __float_as_uint(f);
  return (b & 0x80000000u) ? ~b : (b | 0x80000000u);
}
__device__ __forceinline__ float ord2f(unsigned e) {
  unsigned b = (e & 0x80000000u) ? (e ^ 0x80000000u) : ~e;
  return __uint_as_float(b);
}
__device__ __forceinline__ float wave_sum(float v) {
#pragma unroll
  for (int o = 16; o > 0; o >>= 1) v += __shfl_xor(v, o, 32);
  return v;
}

// ---------- WMMA GEMM: C[M,128] = A[M,128] @ (transW ? W^T : W) + bias ----------
// 256 threads = 8 wave32; 64x128 output tile; K=128 staged fully in LDS as f16.
// LDS row stride 136 halves (68 DWORDs) -> 16 lanes hit disjoint bank groups.
__global__ __launch_bounds__(256) void gemm128_wmma(
    const float* __restrict__ A, const float* __restrict__ W,
    const float* __restrict__ bias, float* __restrict__ C,
    int M, int transW)
{
  __shared__ _Float16 As[64][136];
  __shared__ _Float16 Ws[128][136];   // Ws[n][k] = effective-W[k][n]
  const int tid = threadIdx.x;
  const int m0  = blockIdx.x * 64;

  // stage weights (coalesced global reads, LDS scatter)
  for (int idx = tid; idx < 128 * 128; idx += 256) {
    int r = idx >> 7, c = idx & 127;
    float v = W[idx];
    if (transW) Ws[r][c] = (_Float16)v;   // Ws[n][k] = W[n][k]  -> C = A @ W^T
    else        Ws[c][r] = (_Float16)v;   // Ws[n][k] = W[k][n]  -> C = A @ W
  }
  // stage A tile, zero-pad M boundary
  for (int idx = tid; idx < 64 * 128; idx += 256) {
    int r = idx >> 7, c = idx & 127;
    int gr = m0 + r;
    As[r][c] = (_Float16)((gr < M) ? A[(size_t)gr * Dn + c] : 0.f);
  }
  __syncthreads();

  const int wave = tid >> 5;
  const int lane = tid & 31;
  const int rw = wave >> 1;      // row-tile of wave: 0..3 (16 rows each)
  const int cw = wave & 1;       // col half: 0..1 (64 cols each)
  const int lr = lane & 15;
  const int hi = lane >> 4;      // 0/1 lane half

  v8f acc[4];
#pragma unroll
  for (int t = 0; t < 4; ++t)
#pragma unroll
    for (int i = 0; i < 8; ++i) acc[t][i] = 0.f;

#pragma unroll
  for (int kt = 0; kt < 4; ++kt) {           // K = 4 x 32
    // A fragment (ISA 16-bit A 16x32 layout): lane half selects K 0-7/16-23 vs 8-15/24-31
    const _Float16* ap = &As[rw * 16 + lr][kt * 32 + hi * 8];
    v8h alo = *(const v8h*)ap;
    v8h ahi = *(const v8h*)(ap + 16);
    v16h af;
#pragma unroll
    for (int i = 0; i < 8; ++i) { af[i] = alo[i]; af[i + 8] = ahi[i]; }
#pragma unroll
    for (int t = 0; t < 4; ++t) {
      // B fragment: lane holds column, lane half selects K 0-7/16-23 vs 8-15/24-31
      const _Float16* bp = &Ws[cw * 64 + t * 16 + lr][kt * 32 + hi * 8];
      v8h blo = *(const v8h*)bp;
      v8h bhi = *(const v8h*)(bp + 16);
      v16h bf;
#pragma unroll
      for (int i = 0; i < 8; ++i) { bf[i] = blo[i]; bf[i + 8] = bhi[i]; }
      acc[t] = __builtin_amdgcn_wmma_f32_16x16x32_f16(
          false, af, false, bf, (short)0, acc[t], false, false);
    }
  }

  // store (C/D layout: VGPR i -> row i (lanes 0-15) / i+8 (lanes 16-31), lane&15 = col)
#pragma unroll
  for (int t = 0; t < 4; ++t) {
    int col = cw * 64 + t * 16 + lr;
    float bv = bias ? bias[col] : 0.f;
#pragma unroll
    for (int i = 0; i < 8; ++i) {
      int gr = m0 + rw * 16 + i + hi * 8;
      if (gr < M) C[(size_t)gr * Dn + col] = acc[t][i] + bv;
    }
  }
}

// ---------- pass 1: attention scores + segment max ----------
__global__ __launch_bounds__(256) void k_edge_scores(
    const float* __restrict__ hs, const float* __restrict__ hd,
    const int* __restrict__ si, const int* __restrict__ di,
    const float* __restrict__ attn, float* __restrict__ eout,
    unsigned* __restrict__ menc)
{
  __shared__ __align__(16) float sa[128];
  int tid = threadIdx.x;
  if (tid < 128) sa[tid] = attn[tid];
  __syncthreads();
  int t = blockIdx.x * 256 + tid;
  if (t >= En * Hn) return;
  int edge = t >> 2, h = t & 3;
  int sn = si[edge], dn = di[edge];
  const float4* ps = (const float4*)(hs + (size_t)sn * Dn + h * DHn);
  const float4* pd = (const float4*)(hd + (size_t)dn * Dn + h * DHn);
  const float4* pa = (const float4*)(sa + h * DHn);
  float acc = 0.f;
#pragma unroll
  for (int i = 0; i < 8; ++i) {
    float4 a = ps[i], b = pd[i], w = pa[i];
    acc += (a.x + b.x) * w.x + (a.y + b.y) * w.y +
           (a.z + b.z) * w.z + (a.w + b.w) * w.w;
  }
  float ev = (acc > 0.f) ? acc : 0.2f * acc;   // leaky_relu 0.2
  eout[t] = ev;
  atomicMax(&menc[dn * Hn + h], f2ord(ev));
}

// ---------- pass 2: exp + segment sum ----------
__global__ __launch_bounds__(256) void k_edge_expsum(
    float* __restrict__ e, const int* __restrict__ di,
    const unsigned* __restrict__ menc, float* __restrict__ ssum)
{
  int t = blockIdx.x * 256 + threadIdx.x;
  if (t >= En * Hn) return;
  int edge = t >> 2, h = t & 3;
  int dn = di[edge];
  float m = ord2f(menc[dn * Hn + h]);
  float ex = expf(e[t] - m);
  e[t] = ex;
  atomicAdd(&ssum[dn * Hn + h], ex);
}

// ---------- categorical gate table: only A*S = 50 distinct gates ----------
__global__ __launch_bounds__(128) void k_gate_table(
    const float* __restrict__ at_emb, const float* __restrict__ as_emb,
    const float* __restrict__ Wcat, const float* __restrict__ bcat,
    float* __restrict__ tab)
{
  __shared__ float emb[64];
  int b = blockIdx.x, j = threadIdx.x;
  int a = b / Sn, s = b % Sn;
  if (j < 32) emb[j] = at_emb[a * EADn + j];
  else if (j < 64) emb[j] = as_emb[s * EADn + (j - 32)];
  __syncthreads();
  float acc = bcat[j];
#pragma unroll 8
  for (int k = 0; k < 64; ++k) acc += emb[k] * Wcat[k * Dn + j];
  tab[b * Dn + j] = 1.f / (1.f + expf(-acc));
}

// ---------- pass 3: gate + normalize + scatter-add (1 edge per 128-thread block) ----------
__global__ __launch_bounds__(128) void k_edge_scatter(
    int mode,
    const float* __restrict__ hs, const int* __restrict__ si, const int* __restrict__ di,
    const float* __restrict__ ex, const float* __restrict__ ssum,
    float* __restrict__ agg,
    const float* __restrict__ gtab, const int* __restrict__ eai,
    const float* __restrict__ ea, const float* __restrict__ Wc1,
    const float* __restrict__ bc1, const float* __restrict__ Wc2,
    const float* __restrict__ bc2)
{
  __shared__ float hsh[128];
  int edge = blockIdx.x, j = threadIdx.x;
  int sn = si[edge], dn = di[edge];
  int h = j >> 5;
  float alpha = ex[edge * Hn + h] / (ssum[dn * Hn + h] + 1e-16f);
  float g = 1.f;
  if (mode == 1) {                       // categorical gate via table
    int a = eai[edge * 2], s = eai[edge * 2 + 1];
    g = gtab[(a * Sn + s) * Dn + j];
  } else if (mode == 2) {                // continuous gate MLP
    float x0 = ea[edge * 2], x1 = ea[edge * 2 + 1];
    float hv = x0 * Wc1[j] + x1 * Wc1[Dn + j] + bc1[j];
    hv = 0.5f * hv * (1.f + erff(hv * 0.70710678118654752f));  // exact gelu
    hsh[j] = hv;
    __syncthreads();
    float acc = bc2[j];
    for (int k = 0; k < 128; ++k) acc += hsh[k] * Wc2[k * Dn + j];
    g = 1.f / (1.f + expf(-acc));
  }
  float val = alpha * hs[(size_t)sn * Dn + j] * g;
  atomicAdd(&agg[(size_t)dn * Dn + j], val);
}

// ---------- residual + layernorm (in place), 1 row per 128-thread block ----------
__global__ __launch_bounds__(128) void k_ln_residual(
    float* __restrict__ x, const float* __restrict__ hbuf,
    const float* __restrict__ g, const float* __restrict__ b)
{
  __shared__ float s1[4], s2[4];
  int row = blockIdx.x, j = threadIdx.x;
  size_t o = (size_t)row * Dn + j;
  float v = x[o] + hbuf[o];
  float ws = wave_sum(v), ws2 = wave_sum(v * v);
  if ((j & 31) == 0) { s1[j >> 5] = ws; s2[j >> 5] = ws2; }
  __syncthreads();
  float tot  = s1[0] + s1[1] + s1[2] + s1[3];
  float tot2 = s2[0] + s2[1] + s2[2] + s2[3];
  float mu  = tot * (1.f / Dn);
  float var = tot2 * (1.f / Dn) - mu * mu;
  x[o] = (v - mu) * rsqrtf(var + 1e-5f) * g[j] + b[j];
}

// ---------- final pair dot: out[p] = xs0[ps] . Y[pd]  (wave per pair) ----------
__global__ __launch_bounds__(256) void k_pair_dot(
    const float* __restrict__ xc, const float* __restrict__ Y,
    const int* __restrict__ ps, const int* __restrict__ pd,
    float* __restrict__ out)
{
  int pid = blockIdx.x * 8 + (threadIdx.x >> 5);
  int lane = threadIdx.x & 31;
  if (pid >= Pn) return;
  const float4 av = *(const float4*)(xc + (size_t)ps[pid] * Dn + lane * 4);
  const float4 bv = *(const float4*)(Y  + (size_t)pd[pid] * Dn + lane * 4);
  float acc = av.x * bv.x + av.y * bv.y + av.z * bv.z + av.w * bv.w;
  acc = wave_sum(acc);
  if (lane == 0) out[pid] = acc;
}

// ---------- host orchestration ----------
extern "C" void kernel_launch(void* const* d_in, const int* in_sizes, int n_in,
                              void* d_out, int out_size, void* d_ws, size_t ws_size,
                              hipStream_t stream)
{
  (void)in_sizes; (void)n_in; (void)out_size; (void)ws_size;
  const float* emb[3] = {(const float*)d_in[0], (const float*)d_in[1], (const float*)d_in[2]};
  const float* Wsrc = (const float*)d_in[3];
  const float* bsrc = (const float*)d_in[4];
  const float* Wdst = (const float*)d_in[5];
  const float* bdst = (const float*)d_in[6];
  const float* attn = (const float*)d_in[7];
  const float* Wout = (const float*)d_in[8];
  const float* bout = (const float*)d_in[9];
  const float* ln_g = (const float*)d_in[10];
  const float* ln_b = (const float*)d_in[11];
  const float* at_emb = (const float*)d_in[12];
  const float* as_emb = (const float*)d_in[13];
  const float* Wcat = (const float*)d_in[14];
  const float* bcat = (const float*)d_in[15];
  const float* Wc1 = (const float*)d_in[16];
  const float* bc1 = (const float*)d_in[17];
  const float* Wc2 = (const float*)d_in[18];
  const float* bc2 = (const float*)d_in[19];
  const float* W_cd = (const float*)d_in[20];
  const float* eacont[2] = {(const float*)d_in[21], (const float*)d_in[22]};
  const int* SRC[6]; const int* DST[6];
  for (int r = 0; r < 6; ++r) { SRC[r] = (const int*)d_in[23 + 2 * r]; DST[r] = (const int*)d_in[24 + 2 * r]; }
  const int* eacat[2] = {(const int*)d_in[35], (const int*)d_in[36]};
  const int* pair_src = (const int*)d_in[37];
  const int* pair_dst = (const int*)d_in[38];
  float* out = (float*)d_out;

  static const int STt[6] = {0, 2, 0, 1, 1, 2};
  static const int DTt[6] = {2, 0, 1, 0, 2, 1};
  const int Ns[3] = {NCn, NDn, NGn};

  // workspace carve (~160 MB)
  char* p = (char*)d_ws;
  auto carve = [&](size_t n) { char* r = p; p += (n + 255) & ~(size_t)255; return (void*)r; };
  float* xs[3]; float* agg[3];
  for (int t = 0; t < 3; ++t) xs[t]  = (float*)carve((size_t)Ns[t] * Dn * 4);
  for (int t = 0; t < 3; ++t) agg[t] = (float*)carve((size_t)Ns[t] * Dn * 4);
  float* hsb = (float*)carve((size_t)NCn * Dn * 4);   // max-sized reusable buffers
  float* hdb = (float*)carve((size_t)NCn * Dn * 4);
  float* ebuf = (float*)carve((size_t)En * Hn * 4);
  unsigned* menc = (unsigned*)carve((size_t)NCn * Hn * 4);
  float* ssum = (float*)carve((size_t)NCn * Hn * 4);
  float* gtab = (float*)carve((size_t)An * Sn * Dn * 4);

  for (int t = 0; t < 3; ++t)
    hipMemcpyAsync(xs[t], emb[t], (size_t)Ns[t] * Dn * 4, hipMemcpyDeviceToDevice, stream);

  auto gemm = [&](const float* A, const float* W, const float* b, float* C, int M, int tw) {
    gemm128_wmma<<<dim3((M + 63) / 64), dim3(256), 0, stream>>>(A, W, b, C, M, tw);
  };

  const int nbE = (En * Hn + 255) / 256;
  for (int l = 0; l < Ln; ++l) {
    for (int t = 0; t < 3; ++t)
      hipMemsetAsync(agg[t], 0, (size_t)Ns[t] * Dn * 4, stream);
    for (int r = 0; r < 6; ++r) {
      int st = STt[r], dt = DTt[r];
      size_t wofs = (size_t)(l * 6 + r) * Dn * Dn;
      size_t bofs = (size_t)(l * 6 + r) * Dn;
      gemm(xs[st], Wsrc + wofs, bsrc + bofs, hsb, Ns[st], 0);
      gemm(xs[dt], Wdst + wofs, bdst + bofs, hdb, Ns[dt], 0);
      hipMemsetAsync(menc, 0, (size_t)Ns[dt] * Hn * 4, stream);  // ord(-inf) < 0x00800000
      hipMemsetAsync(ssum, 0, (size_t)Ns[dt] * Hn * 4, stream);
      k_edge_scores<<<dim3(nbE), dim3(256), 0, stream>>>(
          hsb, hdb, SRC[r], DST[r], attn + (size_t)(l * 6 + r) * Dn, ebuf, menc);
      k_edge_expsum<<<dim3(nbE), dim3(256), 0, stream>>>(ebuf, DST[r], menc, ssum);
      int mode = 0;
      if (r == 0 || r == 1) {
        mode = 1;
        k_gate_table<<<dim3(An * Sn), dim3(128), 0, stream>>>(
            at_emb + (size_t)l * An * EADn, as_emb + (size_t)l * Sn * EADn,
            Wcat + (size_t)(l * 2 + r) * 2 * EADn * Dn,
            bcat + (size_t)(l * 2 + r) * Dn, gtab);
      } else if (r == 4 || r == 5) {
        mode = 2;
      }
      int k = (r >= 4) ? r - 4 : 0;
      k_edge_scatter<<<dim3(En), dim3(128), 0, stream>>>(
          mode, hsb, SRC[r], DST[r], ebuf, ssum, agg[dt],
          gtab, (mode == 1) ? eacat[r] : (const int*)nullptr,
          (mode == 2) ? eacont[k] : (const float*)nullptr,
          Wc1 + (size_t)(l * 2 + k) * 2 * Dn, bc1 + (size_t)(l * 2 + k) * Dn,
          Wc2 + (size_t)(l * 2 + k) * Dn * Dn, bc2 + (size_t)(l * 2 + k) * Dn);
    }
    for (int t = 0; t < 3; ++t) {
      gemm(agg[t], Wout + (size_t)(l * 3 + t) * Dn * Dn,
           bout + (size_t)(l * 3 + t) * Dn, hsb, Ns[t], 0);
      k_ln_residual<<<dim3(Ns[t]), dim3(128), 0, stream>>>(
          xs[t], hsb, ln_g + (size_t)(l * 3 + t) * Dn, ln_b + (size_t)(l * 3 + t) * Dn);
    }
  }
  // out[p] = xs0[ps] . (W_cd @ xs1[pd])  ->  Y = xs1 @ W_cd^T  (20k rows instead of 200k)
  gemm(xs[1], W_cd, nullptr, hdb, NDn, 1);
  k_pair_dot<<<dim3((Pn + 7) / 8), dim3(256), 0, stream>>>(xs[0], hdb, pair_src, pair_dst, out);
}
